// SelfAttention_29437705847499
// MI455X (gfx1250) — matrix operand order repaired and checked by reading, not compiled, not run on hardware
//
#include <hip/hip_runtime.h>
#include <hip/hip_bf16.h>

// ---------------------------------------------------------------------------
// Self-attention forward for MI455X (gfx1250), bf16 WMMA + fp32 softmax.
//   B=2, N=2048, C=1024, H=16, D=64.  SCALE = 1/sqrt(64) = 0.125
// ---------------------------------------------------------------------------

#define B_SZ     2
#define N_SEQ    2048
#define C_DIM    1024
#define H_HEADS  16
#define D_HEAD   64
#define TOKENS   (B_SZ * N_SEQ)          // 4096
#define SCALE_F  0.125f

typedef __attribute__((ext_vector_type(16))) __bf16 v16bf;
typedef __attribute__((ext_vector_type(8)))  float  v8f;

__device__ __forceinline__ unsigned short f32_to_bf16(float f) {
  union { float f; unsigned int u; } v; v.f = f;
  unsigned int lsb = (v.u >> 16) & 1u;
  return (unsigned short)((v.u + 0x7FFFu + lsb) >> 16);   // round-nearest-even
}

__device__ __forceinline__ v16bf make_v16(uint4 lo, uint4 hi) {
  union { uint4 q[2]; v16bf v; } u;
  u.q[0] = lo; u.q[1] = hi;
  return u.v;
}

__device__ __forceinline__ v8f wmma_bf16(v16bf a, v16bf b, v8f c) {
  // D(f32 16x16) = A(bf16 16x32) * B(bf16 32x16) + C
  return __builtin_amdgcn_wmma_f32_16x16x32_bf16(
      /*neg_a=*/false, a, /*neg_b=*/false, b,
      /*c_mod=*/(short)0, c, /*reuse_a=*/false, /*reuse_b=*/false);
}

// A-matrix 16x32 bf16 tile: lane l -> row m=l&15, half h=l>>4.
// VGPR0-3 hold K = h*8 .. h*8+7 (16B contiguous), VGPR4-7 hold K = 16+h*8 ..
__device__ __forceinline__ v16bf load_a_tile(const unsigned short* row_ptr, int h) {
  uint4 lo = *(const uint4*)(row_ptr + h * 8);
  uint4 hi = *(const uint4*)(row_ptr + 16 + h * 8);
  return make_v16(lo, hi);
}

// B-matrix 32x16 bf16 tile from a "transposed" source srcT where
// B[k][n] = srcT[n][k] (row-major, stride ld).  Lane l -> col n=l&15, h=l>>4.
// VGPR0-3 hold K = h*8 .. h*8+7, VGPR4-7 hold K = 16+h*8 .. (same as A).
__device__ __forceinline__ v16bf load_b_tile(const unsigned short* col_ptr_k0, int h) {
  uint4 lo = *(const uint4*)(col_ptr_k0 + h * 8);
  uint4 hi = *(const uint4*)(col_ptr_k0 + 16 + h * 8);
  return make_v16(lo, hi);
}

// ---------------------------------------------------------------------------
// Stage 0: f32 -> bf16 conversion
// ---------------------------------------------------------------------------
__global__ void k_cvt_bf16(const float* __restrict__ in,
                           unsigned short* __restrict__ out, int n) {
  int i = blockIdx.x * blockDim.x + threadIdx.x;
  if (i < n) out[i] = f32_to_bf16(in[i]);
}

// ---------------------------------------------------------------------------
// Stage 1: fused QKV projection.  Y = X @ W^T + b  for W in {Wq,Wk,Wv}.
// Each wave -> one 16(M) x 64(N) output tile; K-loop over 1024 in steps of 32.
// Q,K stored bf16 [B,H,N,D]; V stored TRANSPOSED bf16 [B,H,D,N].
// ---------------------------------------------------------------------------
__global__ void k_proj_qkv(const unsigned short* __restrict__ Xbf,
                           const unsigned short* __restrict__ Wqb,
                           const unsigned short* __restrict__ Wkb,
                           const unsigned short* __restrict__ Wvb,
                           const float* __restrict__ bq,
                           const float* __restrict__ bk,
                           const float* __restrict__ bv,
                           unsigned short* __restrict__ Qo,
                           unsigned short* __restrict__ Ko,
                           unsigned short* __restrict__ Vto) {
  const int wave = threadIdx.x >> 5;
  const int lane = threadIdx.x & 31;
  const int gw   = blockIdx.x * 4 + wave;          // 256 * 48 = 12288 waves
  const int nb   = gw % 48;                        // 64-col tile: 0..47 (3*1024/64)
  const int mb   = gw / 48;                        // 16-row tile: 0..255
  const int m = lane & 15, h = lane >> 4;
  const int row0 = mb * 16;

  v8f acc[4] = {v8f{}, v8f{}, v8f{}, v8f{}};

  for (int k0 = 0; k0 < C_DIM; k0 += 32) {
    const unsigned short* pa = Xbf + (size_t)(row0 + m) * C_DIM + k0;
    v16bf a = load_a_tile(pa, h);
#pragma unroll
    for (int t = 0; t < 4; ++t) {
      const int cg    = nb * 64 + t * 16 + m;      // global out column for lane col
      const int which = cg >> 10;                  // 0=q 1=k 2=v
      const int crow  = cg & 1023;
      const unsigned short* W = (which == 0) ? Wqb : (which == 1) ? Wkb : Wvb;
      v16bf b = load_b_tile(W + (size_t)crow * C_DIM + k0, h);
      acc[t] = wmma_bf16(a, b, acc[t]);
    }
  }

  // epilogue: bias + scatter to head layouts
#pragma unroll
  for (int t = 0; t < 4; ++t) {
    const int cg    = nb * 64 + t * 16 + (lane & 15);
    const int which = cg >> 10;
    const int cc    = cg & 1023;
    const float bias = (which == 0) ? bq[cc] : (which == 1) ? bk[cc] : bv[cc];
    const int hh = cc >> 6, d = cc & 63;
#pragma unroll
    for (int r = 0; r < 8; ++r) {
      const int rowm = r + 8 * h;                  // C/D layout row
      const int tok  = row0 + rowm;                // 0..4095
      const int bb   = tok >> 11, nn = tok & 2047;
      const unsigned short bfv = f32_to_bf16(acc[t][r] + bias);
      const size_t bh = (size_t)(bb * H_HEADS + hh);
      if (which == 0)      Qo [(bh * N_SEQ + nn) * D_HEAD + d] = bfv;
      else if (which == 1) Ko [(bh * N_SEQ + nn) * D_HEAD + d] = bfv;
      else                 Vto[(bh * D_HEAD + d) * N_SEQ + nn] = bfv;   // transposed
    }
  }
}

// ---------------------------------------------------------------------------
// Stage 2: flash attention.  One wave = 16 query rows of one (b,h); streams
// 32 keys / iteration.  fp32 online softmax, P repacked C->A layout via LDS.
// Output written bf16 to [B,N,C] staging for the final projection.
// ---------------------------------------------------------------------------
__global__ void k_attn(const unsigned short* __restrict__ Q,
                       const unsigned short* __restrict__ K,
                       const unsigned short* __restrict__ Vt,
                       unsigned short* __restrict__ Aout) {
  __shared__ __align__(16) unsigned short lds_p[4][16 * 32];  // per-wave P tile

  const int wave = threadIdx.x >> 5;
  const int lane = threadIdx.x & 31;
  const int gw   = blockIdx.x * 4 + wave;          // 32 * 128 = 4096 waves
  const int qt   = gw & 127;                       // query tile 0..127
  const int bh   = gw >> 7;                        // (b*16+h) 0..31

  const unsigned short* Qbh = Q  + (size_t)bh * N_SEQ * D_HEAD;
  const unsigned short* Kbh = K  + (size_t)bh * N_SEQ * D_HEAD;
  const unsigned short* Vbh = Vt + (size_t)bh * D_HEAD * N_SEQ;

  const int m = lane & 15, h = lane >> 4;
  const int row0 = qt * 16;

  // Q as two A tiles (d = 0..31, 32..63), hoisted out of the key loop
  v16bf aq[2];
#pragma unroll
  for (int c = 0; c < 2; ++c)
    aq[c] = load_a_tile(Qbh + (size_t)(row0 + m) * D_HEAD + c * 32, h);

  float mi[8], li[8];
#pragma unroll
  for (int r = 0; r < 8; ++r) { mi[r] = -1e30f; li[r] = 0.0f; }
  v8f acc[4] = {v8f{}, v8f{}, v8f{}, v8f{}};      // O tiles over d (4 x 16)

  unsigned short* myp = &lds_p[wave][0];

  for (int kt = 0; kt < N_SEQ; kt += 32) {
    // S = Q * K^T for 32 keys (two 16-col score tiles, K-dim 64 = 2x32)
    v8f s0 = {}, s1 = {};
#pragma unroll
    for (int c = 0; c < 2; ++c) {
      v16bf b0 = load_b_tile(Kbh + (size_t)(kt + m) * D_HEAD + c * 32, h);
      s0 = wmma_bf16(aq[c], b0, s0);
      v16bf b1 = load_b_tile(Kbh + (size_t)(kt + 16 + m) * D_HEAD + c * 32, h);
      s1 = wmma_bf16(aq[c], b1, s1);
    }

    // fp32 online softmax (row m lives in the 16 lanes of one half)
#pragma unroll
    for (int r = 0; r < 8; ++r) {
      float x0 = s0[r] * SCALE_F, x1 = s1[r] * SCALE_F;
      float mx = fmaxf(x0, x1);
      mx = fmaxf(mx, __shfl_xor(mx, 1, 32));
      mx = fmaxf(mx, __shfl_xor(mx, 2, 32));
      mx = fmaxf(mx, __shfl_xor(mx, 4, 32));
      mx = fmaxf(mx, __shfl_xor(mx, 8, 32));
      const float mnew  = fmaxf(mi[r], mx);
      const float alpha = __expf(mi[r] - mnew);
      const float p0 = __expf(x0 - mnew);
      const float p1 = __expf(x1 - mnew);
      float rs = p0 + p1;
      rs += __shfl_xor(rs, 1, 32);
      rs += __shfl_xor(rs, 2, 32);
      rs += __shfl_xor(rs, 4, 32);
      rs += __shfl_xor(rs, 8, 32);
      li[r] = li[r] * alpha + rs;
      mi[r] = mnew;
#pragma unroll
      for (int t = 0; t < 4; ++t) acc[t][r] *= alpha;
      // stash P (C-layout coords) into LDS
      const int rowm = r + 8 * h;
      const int nl   = lane & 15;
      myp[rowm * 32 + nl]      = f32_to_bf16(p0);
      myp[rowm * 32 + 16 + nl] = f32_to_bf16(p1);
    }

    // explicit DS drain before the layout-changing re-read (in-order per wave,
    // compiler also tracks this; extra wait is always safe)
    asm volatile("s_wait_dscnt 0" ::: "memory");

    // re-read P as a 16x32 A tile
    v16bf pa = load_a_tile(myp + m * 32, h);

    // O += P * V   (V transposed: B[k=key][n=d] = Vt[d][key], contiguous)
#pragma unroll
    for (int t = 0; t < 4; ++t) {
      v16bf bv = load_b_tile(Vbh + (size_t)(t * 16 + m) * N_SEQ + kt, h);
      acc[t] = wmma_bf16(pa, bv, acc[t]);
    }
  }

  // epilogue: normalize, write bf16 [B,N,C] staging
  const int bb = bh >> 4, hh = bh & 15;
#pragma unroll
  for (int r = 0; r < 8; ++r) {
    const float inv  = 1.0f / li[r];
    const int rowm   = r + 8 * h;
    const size_t tok = (size_t)bb * N_SEQ + row0 + rowm;
#pragma unroll
    for (int t = 0; t < 4; ++t) {
      const int d = t * 16 + (lane & 15);
      Aout[tok * C_DIM + hh * D_HEAD + d] = f32_to_bf16(acc[t][r] * inv);
    }
  }
}

// ---------------------------------------------------------------------------
// Stage 3: output projection.  out = Attn @ Wo^T + bo  (f32 result)
// ---------------------------------------------------------------------------
__global__ void k_proj_out(const unsigned short* __restrict__ Abf,
                           const unsigned short* __restrict__ Wob,
                           const float* __restrict__ bo,
                           float* __restrict__ out) {
  const int wave = threadIdx.x >> 5;
  const int lane = threadIdx.x & 31;
  const int gw   = blockIdx.x * 4 + wave;          // 256 * 16 = 4096 waves
  const int nb   = gw & 15;                        // 64-col tile
  const int mb   = gw >> 4;                        // 16-row tile
  const int m = lane & 15, h = lane >> 4;
  const int row0 = mb * 16;

  v8f acc[4] = {v8f{}, v8f{}, v8f{}, v8f{}};

  for (int k0 = 0; k0 < C_DIM; k0 += 32) {
    v16bf a = load_a_tile(Abf + (size_t)(row0 + m) * C_DIM + k0, h);
#pragma unroll
    for (int t = 0; t < 4; ++t) {
      const int cg = nb * 64 + t * 16 + m;
      v16bf b = load_b_tile(Wob + (size_t)cg * C_DIM + k0, h);
      acc[t] = wmma_bf16(a, b, acc[t]);
    }
  }

#pragma unroll
  for (int t = 0; t < 4; ++t) {
    const int cg = nb * 64 + t * 16 + (lane & 15);
    const float bias = bo[cg];
#pragma unroll
    for (int r = 0; r < 8; ++r) {
      const int rowm = r + 8 * h;
      out[(size_t)(row0 + rowm) * C_DIM + cg] = acc[t][r] + bias;
    }
  }
}

// ---------------------------------------------------------------------------
// Host launcher
// ---------------------------------------------------------------------------
extern "C" void kernel_launch(void* const* d_in, const int* in_sizes, int n_in,
                              void* d_out, int out_size, void* d_ws, size_t ws_size,
                              hipStream_t stream) {
  const float* x  = (const float*)d_in[0];
  const float* Wq = (const float*)d_in[1];
  const float* bq = (const float*)d_in[2];
  const float* Wk = (const float*)d_in[3];
  const float* bk = (const float*)d_in[4];
  const float* Wv = (const float*)d_in[5];
  const float* bv = (const float*)d_in[6];
  const float* Wo = (const float*)d_in[7];
  const float* bo = (const float*)d_in[8];

  const size_t XN = (size_t)TOKENS * C_DIM;   // 4,194,304
  const size_t WN = (size_t)C_DIM * C_DIM;    // 1,048,576
  const size_t HN = (size_t)B_SZ * H_HEADS * N_SEQ * D_HEAD;  // 4,194,304

  unsigned short* ws  = (unsigned short*)d_ws;
  unsigned short* Xbf = ws;
  unsigned short* Wqb = Xbf + XN;
  unsigned short* Wkb = Wqb + WN;
  unsigned short* Wvb = Wkb + WN;
  unsigned short* Wob = Wvb + WN;
  unsigned short* Qb  = Wob + WN;
  unsigned short* Kb  = Qb + HN;
  unsigned short* Vtb = Kb + HN;
  unsigned short* Abf = Vtb + HN;

  // Stage 0: bf16 conversions
  k_cvt_bf16<<<(int)(XN / 256), 256, 0, stream>>>(x,  Xbf, (int)XN);
  k_cvt_bf16<<<(int)(WN / 256), 256, 0, stream>>>(Wq, Wqb, (int)WN);
  k_cvt_bf16<<<(int)(WN / 256), 256, 0, stream>>>(Wk, Wkb, (int)WN);
  k_cvt_bf16<<<(int)(WN / 256), 256, 0, stream>>>(Wv, Wvb, (int)WN);
  k_cvt_bf16<<<(int)(WN / 256), 256, 0, stream>>>(Wo, Wob, (int)WN);

  // Stage 1: fused QKV projection (12288 waves / 4 per block)
  k_proj_qkv<<<3072, 128, 0, stream>>>(Xbf, Wqb, Wkb, Wvb, bq, bk, bv, Qb, Kb, Vtb);

  // Stage 2: flash attention (4096 waves / 4 per block)
  k_attn<<<1024, 128, 0, stream>>>(Qb, Kb, Vtb, Abf);

  // Stage 3: output projection (4096 waves / 4 per block)
  k_proj_out<<<1024, 128, 0, stream>>>(Abf, Wob, bo, (float*)d_out);
}
